// GCN_47631187312871
// MI455X (gfx1250) — compile-verified, matrix operand order checked by reading, not run on hardware
//
#include <hip/hip_runtime.h>
#include <hip/hip_bf16.h>

// ---------------------------------------------------------------------------
// GCN autoencoder on gfx1250 (MI455X, wave32, WMMA).
//   h_pre = x @ W1                         (WMMA f16->f32)
//   agg1  = b1 + dinv^2*h_pre + scatter(coef*h_pre[src] -> dst)
//   h2    = relu(agg1) @ W2                (WMMA, relu fused into A load)
//   z     = b2 + dinv^2*h2  + scatter(coef*h2[src] -> dst)   (written to d_out)
//   x_rec = z @ Wd + bd                    (WMMA, bias fused into store)
//
// GEMMs: persistent blocks; each wave holds its B fragments in registers for
// the whole kernel and grid-strides over A tiles staged in LDS (b128 in,
// b128 out), then issues the whole WMMA chain after one dscnt wait.
// ---------------------------------------------------------------------------

typedef __attribute__((ext_vector_type(16))) _Float16 v16h;
typedef __attribute__((ext_vector_type(8)))  float    v8f;

union Frag {
    v16h v;
    uint4 q[2];
    unsigned u[8];
};

__device__ __forceinline__ unsigned pack_f16x2(float f0, float f1) {
    _Float16 h0 = (_Float16)f0, h1 = (_Float16)f1;
    unsigned short s0 = __builtin_bit_cast(unsigned short, h0);
    unsigned short s1 = __builtin_bit_cast(unsigned short, h1);
    return (unsigned)s0 | ((unsigned)s1 << 16);
}

// ------------------------- degree / norm kernels ---------------------------

__global__ void k_deg_init(float* __restrict__ deg, int n) {
    int i = blockIdx.x * blockDim.x + threadIdx.x;
    if (i < n) deg[i] = 1.0f;  // +1 for self loop
}

__global__ void k_deg_count(const long long* __restrict__ ei, float* __restrict__ deg,
                            int nE) {
    int e = blockIdx.x * blockDim.x + threadIdx.x;
    if (e < nE) {
        long long d = ei[(long long)nE + e];  // dst row
        atomicAdd(&deg[d], 1.0f);
    }
}

__global__ void k_dinv(float* __restrict__ deg, int n) {
    int i = blockIdx.x * blockDim.x + threadIdx.x;
    if (i < n) deg[i] = rsqrtf(deg[i]);  // in place: deg -> deg^-1/2
}

// --------------------- B-matrix fragment pre-packing -----------------------
// Packs row-major W[K x N] (f32) into the per-lane VGPR order of the
// v_wmma_f32_16x16x32_f16 B operand: out[((ct*KT+kt)*32+lane)*8 + r] holds the
// f16 pair (k,k+1) with k = kt*32 + (r<4 ? 2r : 16+2(r-4)) + (lane>=16 ? 8:0),
// n = ct*16 + (lane&15).  The GEMM loads each B fragment as 2x b128, once.

__global__ void k_packB(const float* __restrict__ W, unsigned* __restrict__ out,
                        int Ktot, int Ntot) {
    int KT = Ktot >> 5, CT = Ntot >> 4;
    int total = CT * KT * 256;
    int idx = blockIdx.x * blockDim.x + threadIdx.x;
    if (idx >= total) return;
    int r    = idx & 7;
    int lane = (idx >> 3) & 31;
    int tile = idx >> 8;
    int kt = tile % KT;
    int ct = tile / KT;
    int hi = lane >> 4;
    int n  = ct * 16 + (lane & 15);
    int kl = (r < 4 ? 2 * r : 16 + 2 * (r - 4)) + hi * 8;
    int k  = kt * 32 + kl;
    float f0 = W[(size_t)k * Ntot + n];
    float f1 = W[(size_t)(k + 1) * Ntot + n];
    out[idx] = pack_f16x2(f0, f1);
}

// ------------------------------ WMMA GEMM ----------------------------------
// C[nRows x Ntot] = op(A[nRows x Ktot]) * B (+ bias).  Persistent blocks of
// 8 waves; wave -> (ct = wave%CT, rowGroup = wave/CT).  B fragments live in
// registers for the whole kernel.

template <int KT, int CT, bool RELU_A, bool ADD_BIAS>
__global__ __launch_bounds__(256) void k_gemm_wmma(
    const float* __restrict__ A, const unsigned* __restrict__ Bp,
    const float* __restrict__ bias, float* __restrict__ C,
    int nRows, int numGroups) {

    constexpr int Ktot = KT * 32;
    constexpr int Ntot = CT * 16;
    constexpr int K2   = Ktot >> 1;   // u32 (f16 pairs) per A row
    constexpr int K8   = Ktot >> 3;   // 8-float chunks per A row
    constexpr int G    = 8 / CT;      // row sub-tiles per block iteration
    constexpr int ROWS = 16 * G;

    __shared__ __align__(16) unsigned ldsA[ROWS * K2];

    const int tid  = threadIdx.x;
    const int wave = tid >> 5;
    const int lane = tid & 31;
    const int ct   = wave % CT;
    const int rg   = wave / CT;
    const int hi   = (lane >> 4) & 1;
    const int arow = lane & 15;

    // ---- preload this wave's B fragments into registers (once) ----
    Frag b[KT];
#pragma unroll
    for (int kt = 0; kt < KT; ++kt) {
        const uint4* bb = (const uint4*)(Bp + (((size_t)(ct * KT + kt)) * 32 + lane) * 8);
        b[kt].q[0] = bb[0];
        b[kt].q[1] = bb[1];
    }
    const int col = ct * 16 + arow;
    float bv = 0.0f;
    if constexpr (ADD_BIAS) bv = bias[col];

    // ---- persistent loop over 16*G-row tile groups ----
    for (int tg = blockIdx.x; tg < numGroups; tg += gridDim.x) {
        const int rowBase0 = tg * ROWS;

        // cooperative stage of the ROWS x Ktot A tile:
        // 8 contiguous f32 per thread -> 2x global_load_b128, 1x ds_store_b128
        for (int it = tid; it < ROWS * K8; it += 256) {
            int r  = it / K8;
            int k8 = it - r * K8;
            int row = rowBase0 + r;
            if (row >= nRows) row = nRows - 1;  // clamp (stores are guarded)
            const float4* ap = (const float4*)(A + (size_t)row * Ktot + 8 * k8);
            float4 f0 = ap[0], f1 = ap[1];
            if constexpr (RELU_A) {
                f0.x = fmaxf(f0.x, 0.f); f0.y = fmaxf(f0.y, 0.f);
                f0.z = fmaxf(f0.z, 0.f); f0.w = fmaxf(f0.w, 0.f);
                f1.x = fmaxf(f1.x, 0.f); f1.y = fmaxf(f1.y, 0.f);
                f1.z = fmaxf(f1.z, 0.f); f1.w = fmaxf(f1.w, 0.f);
            }
            uint4 pk;
            pk.x = pack_f16x2(f0.x, f0.y);
            pk.y = pack_f16x2(f0.z, f0.w);
            pk.z = pack_f16x2(f1.x, f1.y);
            pk.w = pack_f16x2(f1.z, f1.w);
            *(uint4*)&ldsA[r * K2 + 4 * k8] = pk;
        }
        __syncthreads();

        // fetch all A fragments, then run the WMMA chain
        Frag a[KT];
        const unsigned* abase = &ldsA[(rg * 16 + arow) * K2 + hi * 4];
#pragma unroll
        for (int kt = 0; kt < KT; ++kt) {
            a[kt].q[0] = *(const uint4*)(abase + kt * 16);
            a[kt].q[1] = *(const uint4*)(abase + kt * 16 + 8);
        }
        v8f acc = {};
#pragma unroll
        for (int kt = 0; kt < KT; ++kt) {
            acc = __builtin_amdgcn_wmma_f32_16x16x32_f16(
                false, a[kt].v, false, b[kt].v, (short)0, acc, false, false);
        }

        // C layout: VGPR r -> row = base + r + 8*hi, col = ct*16 + (lane&15)
        const int rowTop = rowBase0 + rg * 16 + hi * 8;
        float* cbase = C + (size_t)rowTop * Ntot + col;
        if (rowBase0 + ROWS <= nRows) {   // fast path: full tile (uniform)
#pragma unroll
            for (int r = 0; r < 8; ++r) {
                float v = acc[r];
                if constexpr (ADD_BIAS) v += bv;
                cbase[(size_t)r * Ntot] = v;
            }
        } else {
#pragma unroll
            for (int r = 0; r < 8; ++r) {
                float v = acc[r];
                if constexpr (ADD_BIAS) v += bv;
                if (rowTop + r < nRows) cbase[(size_t)r * Ntot] = v;
            }
        }
        __syncthreads();  // before overwriting ldsA next iteration
    }
}

// ---------------- aggregation: self-loop init + edge scatter ---------------

__global__ void k_selfinit(const float* __restrict__ hpre, const float* __restrict__ dinv,
                           const float* __restrict__ bias, float* __restrict__ out,
                           int n, int F) {
    long long idx = (long long)blockIdx.x * blockDim.x + threadIdx.x;
    if (idx >= (long long)n * F) return;
    int node = (int)(idx / F);
    int f    = (int)(idx - (long long)node * F);
    float di = dinv[node];
    out[idx] = hpre[idx] * di * di + bias[f];
}

__global__ void k_scatter(const long long* __restrict__ ei, const float* __restrict__ dinv,
                          const float* __restrict__ feat, float* __restrict__ agg,
                          int nE, int F) {
    const int chunks = F >> 2;  // float4 chunks per feature row
    long long t = (long long)blockIdx.x * blockDim.x + threadIdx.x;
    if (t >= (long long)nE * chunks) return;
    int e = (int)(t / chunks);
    int c = (int)(t - (long long)e * chunks);
    long long s = ei[e];
    long long d = ei[(long long)nE + e];
    float coef = dinv[s] * dinv[d];
    const float4 v = *(const float4*)(feat + (size_t)s * F + 4 * c);
    float* o = agg + (size_t)d * F + 4 * c;
    atomicAdd(o + 0, v.x * coef);
    atomicAdd(o + 1, v.y * coef);
    atomicAdd(o + 2, v.z * coef);
    atomicAdd(o + 3, v.w * coef);
}

// ------------------------------- launcher ----------------------------------

static inline int igrid(long long work, int tpb) {
    return (int)((work + tpb - 1) / tpb);
}

extern "C" void kernel_launch(void* const* d_in, const int* in_sizes, int n_in,
                              void* d_out, int out_size, void* d_ws, size_t ws_size,
                              hipStream_t stream) {
    (void)n_in; (void)out_size; (void)ws_size;
    const int IN = 128, HID = 128, OUT = 64;
    const int N = in_sizes[0] / IN;
    const int E = in_sizes[1] / 2;

    const float*     x  = (const float*)d_in[0];
    const long long* ei = (const long long*)d_in[1];
    const float*     W1 = (const float*)d_in[2];
    const float*     b1 = (const float*)d_in[3];
    const float*     W2 = (const float*)d_in[4];
    const float*     b2 = (const float*)d_in[5];
    const float*     Wd = (const float*)d_in[6];
    const float*     bd = (const float*)d_in[7];

    float* x_rec = (float*)d_out;                     // [N, IN]
    float* z     = (float*)d_out + (size_t)N * IN;    // [N, OUT]

    // workspace carve-out
    float* dinv = (float*)d_ws;                       // N      (deg -> dinv)
    float* hpre = dinv + N;                           // N*HID
    float* agg1 = hpre + (size_t)N * HID;             // N*HID
    float* h2   = agg1 + (size_t)N * HID;             // N*OUT
    unsigned* w1p = (unsigned*)(h2 + (size_t)N * OUT);           // 8*4*256
    unsigned* w2p = w1p + (HID / 16) * (IN / 32) * 256;          // 4*4*256
    unsigned* wdp = w2p + (OUT / 16) * (HID / 32) * 256;         // 8*2*256

    const int T = 256;
    // 1) symmetric-normalization coefficients
    k_deg_init<<<igrid(N, T), T, 0, stream>>>(dinv, N);
    k_deg_count<<<igrid(E, T), T, 0, stream>>>(ei, dinv, E);
    k_dinv<<<igrid(N, T), T, 0, stream>>>(dinv, N);

    // 2) pre-pack weight fragments for the WMMA B-operand layout
    k_packB<<<igrid((IN / 32) * (HID / 16) * 256, T), T, 0, stream>>>(W1, w1p, IN, HID);
    k_packB<<<igrid((HID / 32) * (OUT / 16) * 256, T), T, 0, stream>>>(W2, w2p, HID, OUT);
    k_packB<<<igrid((OUT / 32) * (IN / 16) * 256, T), T, 0, stream>>>(Wd, wdp, OUT, IN);

    const int rowTiles = (N + 15) / 16;
    const int PERSIST  = 1024;  // persistent blocks for the WMMA GEMMs

    // 3) conv1: h_pre = x @ W1 ; agg1 = b1 + dinv^2*h_pre + scatter
    {
        int groups = rowTiles;  // G = 1
        k_gemm_wmma<4, 8, false, false>
            <<<(groups < PERSIST ? groups : PERSIST), 256, 0, stream>>>(
                x, w1p, nullptr, hpre, N, groups);
        k_selfinit<<<igrid((long long)N * HID, T), T, 0, stream>>>(hpre, dinv, b1, agg1, N, HID);
        k_scatter<<<igrid((long long)E * (HID / 4), T), T, 0, stream>>>(ei, dinv, hpre, agg1, E, HID);
    }

    // 4) conv2: h2 = relu(agg1) @ W2 ; z = b2 + dinv^2*h2 + scatter (in d_out)
    {
        int groups = (rowTiles + 1) / 2;  // G = 2
        k_gemm_wmma<4, 4, true, false>
            <<<(groups < PERSIST ? groups : PERSIST), 256, 0, stream>>>(
                agg1, w2p, nullptr, h2, N, groups);
        k_selfinit<<<igrid((long long)N * OUT, T), T, 0, stream>>>(h2, dinv, b2, z, N, OUT);
        k_scatter<<<igrid((long long)E * (OUT / 4), T), T, 0, stream>>>(ei, dinv, h2, z, E, OUT);
    }

    // 5) decoder: x_rec = z @ Wd + bd
    {
        int groups = rowTiles;  // G = 1
        k_gemm_wmma<2, 8, false, true>
            <<<(groups < PERSIST ? groups : PERSIST), 256, 0, stream>>>(
                z, wdp, bd, x_rec, N, groups);
    }
}
